// SubLSTM_6708738916593
// MI455X (gfx1250) — matrix operand order repaired and checked by reading, not compiled, not run on hardware
//
#include <hip/hip_runtime.h>
#include <hip/hip_bf16.h>

// ---------------------------------------------------------------------------
// SubLSTM: B=32, T=1024, I=H=768, G=4H=3072
//   Phase 1: xg = x @ w_ih^T + (b_ih+b_hh)  (bf16 WMMA GEMM, 2x4 reg-blocked)
//   Phase 2: persistent scan, 48 WGs; per-step [32x768]@[768x64] bf16 WMMA.
//            TDM DMA's BOTH the h state (with pad-swizzled LDS pitch) and the
//            xg gate slice into LDS; xg transfer overlaps the WMMA loop.
//            Device-wide atomic barrier between timesteps.
// ---------------------------------------------------------------------------

typedef __attribute__((ext_vector_type(16))) __bf16 bf16x16;
typedef __attribute__((ext_vector_type(8)))  __bf16 bf16x8;
typedef __attribute__((ext_vector_type(8)))  float  f32x8;
typedef __attribute__((ext_vector_type(4)))  unsigned int u32x4;
typedef __attribute__((ext_vector_type(8)))  int          i32x8;
typedef __attribute__((ext_vector_type(4)))  int          i32x4;

#define B_      32
#define T_      1024
#define I_      768
#define H_      768
#define G_      3072     // 4*H
#define NWG     48       // scan workgroups; each owns H_/NWG = 16 h-cols
#define HPITCH  792      // bf16 elems per h_lds row: 1536B data + 3x16B TDM pads

#if defined(__AMDGCN__) && __has_builtin(__builtin_amdgcn_tensor_load_to_lds)
#define HAVE_TDM 1
#else
#define HAVE_TDM 0
#endif

#if defined(__AMDGCN__) && __has_builtin(__builtin_amdgcn_s_wait_tensorcnt)
#define WAIT_TENSORCNT(n) __builtin_amdgcn_s_wait_tensorcnt(n)
#elif defined(__AMDGCN__)
#define WAIT_TENSORCNT(n) asm volatile("s_wait_tensorcnt %0" :: "i"(n) : "memory")
#else
#define WAIT_TENSORCNT(n) ((void)0)
#endif

// TDM builtin arity differs across toolchains: clang-22 (ROCm 7.2) = 5 args,
// clang-23+ (amdgpu-toolchain / therock) = 6 args (extra i32x8 before cpol).
#if HAVE_TDM
#if __clang_major__ >= 23
#define TDM_LOAD(g0, g1, g2, g3) \
    do { i32x8 g4_ = {};         \
         __builtin_amdgcn_tensor_load_to_lds((g0), (g1), (g2), (g3), g4_, 0); } while (0)
#else
#define TDM_LOAD(g0, g1, g2, g3) \
    __builtin_amdgcn_tensor_load_to_lds((g0), (g1), (g2), (g3), 0)
#endif
#endif

__device__ __forceinline__ float sigmoidf_(float x) {
    return 1.0f / (1.0f + __expf(-x));
}

// A fragment (16x32 bf16, MxK), dense pitch. ISA 7.12.2: lane L -> row L%16;
// elems 0..7 -> K = k0 + half*8 + e; elems 8..15 -> K = k0 + 16 + half*8 + e.
__device__ __forceinline__ bf16x16 load_frag_a(const __bf16* p, int pitch,
                                               int row0, int k0, int lane) {
    int r = row0 + (lane & 15);
    int half = lane >> 4;
    union { bf16x16 v; bf16x8 h[2]; } u;
    const __bf16* base = p + r * pitch + k0;
    u.h[0] = *(const bf16x8*)(base + half * 8);
    u.h[1] = *(const bf16x8*)(base + 16 + half * 8);
    return u.v;
}

// A fragment from the pad-swizzled h_lds: 16B pad after every 512B of data,
// i.e. +8 elements per 256-element K segment (blocks never straddle pads).
__device__ __forceinline__ bf16x16 load_frag_a_pad(const __bf16* hb,
                                                   int row0, int k0, int lane) {
    int r = row0 + (lane & 15);
    int half = lane >> 4;
    int kadj = k0 + ((k0 >> 8) << 3);
    union { bf16x16 v; bf16x8 h[2]; } u;
    const __bf16* base = hb + r * HPITCH + kadj;
    u.h[0] = *(const bf16x8*)(base + half * 8);
    u.h[1] = *(const bf16x8*)(base + 16 + half * 8);
    return u.v;
}

// B fragment (32x16 bf16, KxN), col n contiguous in K at p[n*pitch + k].
// lane L -> col N = L%16; elems 0..7 -> K = k0 + half*8 + e; elems 8..15 -> K = k0 + 16 + half*8 + e.
__device__ __forceinline__ bf16x16 load_frag_b(const __bf16* p, int pitch,
                                               int col0, int k0, int lane) {
    int c = col0 + (lane & 15);
    int half = lane >> 4;
    union { bf16x16 v; bf16x8 h[2]; } u;
    const __bf16* base = p + c * pitch + k0;
    u.h[0] = *(const bf16x8*)(base + half * 8);
    u.h[1] = *(const bf16x8*)(base + 16 + half * 8);
    return u.v;
}

// ---------------------------------------------------------------------------
__global__ void init_kernel(unsigned* bar, __bf16* hbuf) {
    int i = blockIdx.x * blockDim.x + threadIdx.x;
    if (i == 0) *bar = 0u;
    if (i < B_ * H_) hbuf[i] = (__bf16)0.0f;   // h0 = 0 (buffer parity 0)
}

__global__ void cvt_kernel(const float* __restrict__ src, __bf16* __restrict__ dst, int n) {
    int i = blockIdx.x * blockDim.x + threadIdx.x;
    if (i < n) dst[i] = (__bf16)src[i];
}

__global__ void bias_kernel(const float* __restrict__ b_ih,
                            const float* __restrict__ b_hh,
                            float* __restrict__ bsum) {
    int i = blockIdx.x * blockDim.x + threadIdx.x;
    if (i < G_) bsum[i] = b_ih[i] + b_hh[i];
}

// ---------------------------------------------------------------------------
// Phase 1: xg[32768,3072] = x_bf16 @ w_ih_bf16^T + bsum.
// Block = 64 rows x 256 cols; 8 waves as 2x4; each wave 32x64 (2x4 accs).
// ---------------------------------------------------------------------------
__global__ void __launch_bounds__(256)
gemm_xg_kernel(const __bf16* __restrict__ xb, const __bf16* __restrict__ wih,
               const float* __restrict__ bsum, float* __restrict__ xg) {
    const int wid  = threadIdx.x >> 5, lane = threadIdx.x & 31;
    const int wm   = wid >> 2, wn = wid & 3;
    const int row0 = blockIdx.x * 64 + wm * 32;
    const int col0 = blockIdx.y * 256 + wn * 64;

    f32x8 acc[2][4] = {};
#pragma unroll 2
    for (int ks = 0; ks < I_; ks += 32) {
        bf16x16 a0 = load_frag_a(xb, I_, row0,      ks, lane);
        bf16x16 a1 = load_frag_a(xb, I_, row0 + 16, ks, lane);
        bf16x16 b0 = load_frag_b(wih, I_, col0,      ks, lane);
        bf16x16 b1 = load_frag_b(wih, I_, col0 + 16, ks, lane);
        bf16x16 b2 = load_frag_b(wih, I_, col0 + 32, ks, lane);
        bf16x16 b3 = load_frag_b(wih, I_, col0 + 48, ks, lane);
        acc[0][0] = __builtin_amdgcn_wmma_f32_16x16x32_bf16(false, a0, false, b0, (short)0, acc[0][0], false, false);
        acc[0][1] = __builtin_amdgcn_wmma_f32_16x16x32_bf16(false, a0, false, b1, (short)0, acc[0][1], false, false);
        acc[0][2] = __builtin_amdgcn_wmma_f32_16x16x32_bf16(false, a0, false, b2, (short)0, acc[0][2], false, false);
        acc[0][3] = __builtin_amdgcn_wmma_f32_16x16x32_bf16(false, a0, false, b3, (short)0, acc[0][3], false, false);
        acc[1][0] = __builtin_amdgcn_wmma_f32_16x16x32_bf16(false, a1, false, b0, (short)0, acc[1][0], false, false);
        acc[1][1] = __builtin_amdgcn_wmma_f32_16x16x32_bf16(false, a1, false, b1, (short)0, acc[1][1], false, false);
        acc[1][2] = __builtin_amdgcn_wmma_f32_16x16x32_bf16(false, a1, false, b2, (short)0, acc[1][2], false, false);
        acc[1][3] = __builtin_amdgcn_wmma_f32_16x16x32_bf16(false, a1, false, b3, (short)0, acc[1][3], false, false);
    }
#pragma unroll
    for (int in = 0; in < 4; ++in) {
        const int n = col0 + in * 16 + (lane & 15);
        const float bia = bsum[n];
#pragma unroll
        for (int im = 0; im < 2; ++im) {
            const int rbase = row0 + im * 16 + 8 * (lane >> 4);
#pragma unroll
            for (int v = 0; v < 8; ++v)
                xg[(size_t)(rbase + v) * G_ + n] = acc[im][in][v] + bia;
        }
    }
}

// ---------------------------------------------------------------------------
// Phase 2: persistent scan.
// ---------------------------------------------------------------------------
__global__ void __launch_bounds__(256)
scan_kernel(const float* __restrict__ xg, const __bf16* __restrict__ whh,
            __bf16* __restrict__ hbuf, float* __restrict__ out,
            unsigned* __restrict__ bar) {
    __shared__ __align__(16) __bf16 h_lds[B_ * HPITCH];  // 50,688 B, pad-swizzled
    __shared__ __align__(16) float  gx_lds[B_][4][16];   // 8 KB: xg tile / gates
    __shared__ float  c_lds[B_][16];                     // 2 KB: cell state

    const int wg   = blockIdx.x;
    const int tid  = threadIdx.x;
    const int wid  = tid >> 5, lane = tid & 31;
    const int mt   = wid >> 2, nt = wid & 3;
    const int colh = wg * 16;
    const int g0c  = nt * H_ + colh;

#if HAVE_TDM
    const unsigned lds_gx = (unsigned)(size_t)(void*)&gx_lds[0][0][0];
    const unsigned lds_h  = (unsigned)(size_t)(void*)&h_lds[0];
    // xg D#: data_size=4B; dim0=16 contig, dim1=4 stride 768, dim2=32 stride T*G.
    const i32x8 tdm_xg_g1 = { 0x20000, 16 << 16, 4 << 16, 16 << 16,
                              4 | (32 << 16), H_, 0, 0x30 };
    const i32x4 tdm_xg_g2 = { 32, 0, 0, 0 };
    // h D#: data_size=2B; 768x32 tile, stride 768; pad_enable, interval=512B
    // (code 6), amount=16B (code 3) -> LDS pitch 1584B, rows at bank r*12%64.
    const i32x8 tdm_h_g1  = { (1 << 16) | (1 << 20) | (6 << 22) | (3 << 25),
                              768 << 16, 32 << 16, 768 << 16, 32, H_, 0, 0 };
    const i32x4 tdm_zero4 = { 0, 0, 0, 0 };
#endif

    for (int i = tid; i < B_ * 16; i += 256)
        c_lds[i >> 4][i & 15] = 0.0f;
    __syncthreads();

    for (int t = 0; t < T_; ++t) {
#if HAVE_TDM
        if (wid == 0) {   // one issue per WG; EXEC ignored for tensor ops
            // --- TDM #1: h state -> padded h_lds ---
            unsigned long long ha = (unsigned long long)(size_t)hbuf
                                  + (unsigned long long)(t & 1) * (B_ * H_ * 2);
            u32x4 g0h;
            g0h.x = 1u;
            g0h.y = lds_h;
            g0h.z = (unsigned)ha;
            g0h.w = ((unsigned)(ha >> 32) & 0x01FFFFFFu) | 0x80000000u;
            TDM_LOAD(g0h, tdm_h_g1, tdm_zero4, tdm_zero4);
            // --- TDM #2: xg gate slice -> gx_lds (overlaps WMMA loop) ---
            unsigned long long ga = (unsigned long long)(size_t)xg
                                  + 4ull * ((unsigned long long)t * G_ + (unsigned)colh);
            u32x4 g0x;
            g0x.x = 1u;
            g0x.y = lds_gx;
            g0x.z = (unsigned)ga;
            g0x.w = ((unsigned)(ga >> 32) & 0x01FFFFFFu) | 0x80000000u;
            TDM_LOAD(g0x, tdm_xg_g1, tdm_xg_g2, tdm_zero4);
            WAIT_TENSORCNT(1);   // in-order: h transfer complete; xg in flight
        }
#else
        // fallback: stage h manually into the padded layout
        {
            const bf16x8* src = (const bf16x8*)(hbuf + (size_t)(t & 1) * B_ * H_);
            for (int ci = tid; ci < (B_ * H_) / 8; ci += 256) {
                int r = ci / (H_ / 8);
                int k = (ci % (H_ / 8)) * 8;
                int kadj = k + ((k >> 8) << 3);
                *(bf16x8*)&h_lds[r * HPITCH + kadj] = src[ci];
            }
        }
        if (t + 1 < T_) {
            int b = tid & 31;
            __builtin_prefetch(&xg[((size_t)b * T_ + (t + 1)) * G_ + g0c], 0, 1);
        }
#endif
        __syncthreads();   // h_lds ready for all waves

        // ---- gates(pre) = h @ w_hh^T slice: [16x768]@[768x16] per wave ----
        // Two interleaved accumulator chains to break the WMMA RAW chain.
        f32x8 acc0 = {}, acc1 = {};
#pragma unroll 4
        for (int ks = 0; ks < H_; ks += 64) {
            bf16x16 a0 = load_frag_a_pad(h_lds, mt * 16, ks, lane);
            bf16x16 b0 = load_frag_b(whh, H_, g0c, ks, lane);
            acc0 = __builtin_amdgcn_wmma_f32_16x16x32_bf16(false, a0, false, b0,
                                                           (short)0, acc0, false, false);
            bf16x16 a1 = load_frag_a_pad(h_lds, mt * 16, ks + 32, lane);
            bf16x16 b1 = load_frag_b(whh, H_, g0c, ks + 32, lane);
            acc1 = __builtin_amdgcn_wmma_f32_16x16x32_bf16(false, a1, false, b1,
                                                           (short)0, acc1, false, false);
        }
        f32x8 acc = acc0 + acc1;

#if HAVE_TDM
        if (wid == 0) WAIT_TENSORCNT(0);   // xg tile landed
        __syncthreads();                   // visible to all waves
#endif

        // ---- += xg, sigmoid; gates written in place into gx_lds ----
        {
            const int n = lane & 15;
            const int rbase = mt * 16 + 8 * (lane >> 4);
#pragma unroll
            for (int v = 0; v < 8; ++v) {
                int r = rbase + v;
#if HAVE_TDM
                float xgv = gx_lds[r][nt][n];
#else
                float xgv = xg[((size_t)r * T_ + t) * G_ + g0c + n];
#endif
                gx_lds[r][nt][n] = sigmoidf_(acc[v] + xgv);
            }
        }
        __syncthreads();

        // ---- cell update: c' = f*c + (z-i); h' = sigmoid(c') - o ----
        for (int i = tid; i < B_ * 16; i += 256) {
            int r = i >> 4, cc = i & 15;
            float ig = gx_lds[r][0][cc];
            float fg = gx_lds[r][1][cc];
            float zg = gx_lds[r][2][cc];
            float og = gx_lds[r][3][cc];
            float c  = fg * c_lds[r][cc] + (zg - ig);
            c_lds[r][cc] = c;
            float h = sigmoidf_(c) - og;
            out[((size_t)r * T_ + t) * H_ + colh + cc] = h;
            hbuf[(size_t)((t + 1) & 1) * B_ * H_ + (size_t)r * H_ + colh + cc] = (__bf16)h;
        }
        __syncthreads();

        // ---- device-wide step barrier ----
        if (tid == 0) {
            __threadfence();
            atomicAdd(bar, 1u);
            const unsigned target = (unsigned)NWG * (unsigned)(t + 1);
            while (__hip_atomic_load(bar, __ATOMIC_ACQUIRE,
                                     __HIP_MEMORY_SCOPE_AGENT) < target) {
                __builtin_amdgcn_s_sleep(1);
            }
        }
        __syncthreads();
    }
}

// ---------------------------------------------------------------------------
extern "C" void kernel_launch(void* const* d_in, const int* in_sizes, int n_in,
                              void* d_out, int out_size, void* d_ws, size_t ws_size,
                              hipStream_t stream) {
    (void)in_sizes; (void)n_in; (void)out_size; (void)ws_size;
    const float* x    = (const float*)d_in[0];
    const float* w_ih = (const float*)d_in[1];
    const float* w_hh = (const float*)d_in[2];
    const float* b_ih = (const float*)d_in[3];
    const float* b_hh = (const float*)d_in[4];
    float* out = (float*)d_out;

    char* ws = (char*)d_ws;
    const size_t OFF_BAR  = 0;
    const size_t OFF_WIH  = 256;
    const size_t OFF_WHH  = OFF_WIH + (size_t)G_ * I_ * 2;
    const size_t OFF_BSUM = OFF_WHH + (size_t)G_ * H_ * 2;
    const size_t OFF_HBUF = OFF_BSUM + (size_t)G_ * 4;
    const size_t OFF_XB   = OFF_HBUF + (size_t)2 * B_ * H_ * 2;
    const size_t OFF_XG   = OFF_XB + (size_t)B_ * T_ * I_ * 2;

    unsigned* bar    = (unsigned*)(ws + OFF_BAR);
    __bf16*   wih_bf = (__bf16*)(ws + OFF_WIH);
    __bf16*   whh_bf = (__bf16*)(ws + OFF_WHH);
    float*    bsum   = (float*) (ws + OFF_BSUM);
    __bf16*   hbuf   = (__bf16*)(ws + OFF_HBUF);
    __bf16*   xb     = (__bf16*)(ws + OFF_XB);
    float*    xg     = (float*) (ws + OFF_XG);

    init_kernel<<<(B_ * H_ + 255) / 256, 256, 0, stream>>>(bar, hbuf);
    cvt_kernel<<<(B_ * T_ * I_ + 255) / 256, 256, 0, stream>>>(x, xb, B_ * T_ * I_);
    cvt_kernel<<<(G_ * I_ + 255) / 256, 256, 0, stream>>>(w_ih, wih_bf, G_ * I_);
    cvt_kernel<<<(G_ * H_ + 255) / 256, 256, 0, stream>>>(w_hh, whh_bf, G_ * H_);
    bias_kernel<<<(G_ + 255) / 256, 256, 0, stream>>>(b_ih, b_hh, bsum);

    dim3 g1(512, 12);
    gemm_xg_kernel<<<g1, 256, 0, stream>>>(xb, wih_bf, bsum, xg);

    scan_kernel<<<NWG, 256, 0, stream>>>(xg, whh_bf, hbuf, out, bar);
}
